// BaseX2HAttLayer_2259152797787
// MI455X (gfx1250) — compile-verified, matrix operand order checked by reading, not run on hardware
//
#include <hip/hip_runtime.h>
#include <math.h>

// ---------------- constants ----------------
#define NN      10000
#define EE      320000
#define DIM     128
#define KVD     324
#define KVPAD   352            // 324 padded to multiple of 32
#define NHEAD   16
#define HDIM    8

typedef _Float16 v16h __attribute__((ext_vector_type(16)));
typedef _Float16 h8   __attribute__((ext_vector_type(8)));
typedef _Float16 h4   __attribute__((ext_vector_type(4)));
typedef float    v8f  __attribute__((ext_vector_type(8)));

union FragH { v16h v; h8 h[2]; };

// ---- float->ordered-unsigned key for atomicMax on floats ----
__device__ __forceinline__ unsigned enc_key(float f) {
  unsigned u = __float_as_uint(f);
  return (u & 0x80000000u) ? ~u : (u | 0x80000000u);
}
__device__ __forceinline__ float dec_key(unsigned k) {
  unsigned u = (k & 0x80000000u) ? (k & 0x7fffffffu) : ~k;
  return __uint_as_float(u);
}

// ---- one 16x16 f32 tile:  C = A(16xK) * B(Kx16)
__device__ __forceinline__ v8f mm_tile(const _Float16* __restrict__ A, int lda,
                                       const _Float16* __restrict__ Bt, int ldb,
                                       int K, int lane) {
  v8f c = {};
  const int r    = lane & 15;
  const int hsel = (lane >> 4) * 8;
  const _Float16* pa = A  + r * lda + hsel;
  const _Float16* pb = Bt + r * ldb + hsel;
  for (int k = 0; k < K; k += 32) {
    FragH fa, fb;
    fa.h[0] = *(const h8*)(pa + k);
    fa.h[1] = *(const h8*)(pa + k + 16);
    fb.h[0] = *(const h8*)(pb + k);
    fb.h[1] = *(const h8*)(pb + k + 16);
    c = __builtin_amdgcn_wmma_f32_16x16x32_f16(false, fa.v, false, fb.v,
                                               (short)0, c, false, false);
  }
  return c;
}

// ---- M=32 blocked: two 16-row A tiles share every B fragment (2x B reuse) ----
__device__ __forceinline__ void mm_tile_m2(const _Float16* __restrict__ A, int lda,
                                           const _Float16* __restrict__ Bt, int ldb,
                                           int K, int lane, v8f& c0, v8f& c1) {
  const int r    = lane & 15;
  const int hsel = (lane >> 4) * 8;
  const _Float16* pa0 = A + r * lda + hsel;
  const _Float16* pa1 = pa0 + 16 * lda;
  const _Float16* pb  = Bt + r * ldb + hsel;
  for (int k = 0; k < K; k += 32) {
    FragH fb, fa0, fa1;
    fb.h[0]  = *(const h8*)(pb + k);
    fb.h[1]  = *(const h8*)(pb + k + 16);
    fa0.h[0] = *(const h8*)(pa0 + k);
    fa0.h[1] = *(const h8*)(pa0 + k + 16);
    fa1.h[0] = *(const h8*)(pa1 + k);
    fa1.h[1] = *(const h8*)(pa1 + k + 16);
    c0 = __builtin_amdgcn_wmma_f32_16x16x32_f16(false, fa0.v, false, fb.v,
                                                (short)0, c0, false, false);
    c1 = __builtin_amdgcn_wmma_f32_16x16x32_f16(false, fa1.v, false, fb.v,
                                                (short)0, c1, false, false);
  }
}

// C/D tile layout: lane holds col n=(lane&15), rows m = mbase + (lane>>4)*8 + r
__device__ __forceinline__ void store_c_bias(float* H, int ldh, int ncol0, int mbase,
                                             v8f c, const float* __restrict__ bias,
                                             int lane) {
  const int n  = ncol0 + (lane & 15);
  const int m0 = mbase + (lane >> 4) * 8;
  const float b = bias[n];
  for (int r = 0; r < 8; ++r) H[(m0 + r) * ldh + n] = c[r] + b;
}

// LayerNorm(128)+ReLU, 16 rows (lanes 0..15, one row each)
__device__ __forceinline__ void ln_relu16(const float* __restrict__ H,
                                          _Float16* __restrict__ A2,
                                          const float* __restrict__ g,
                                          const float* __restrict__ be, int lane) {
  if (lane < 16) {
    const float* row = H + lane * DIM;
    float mu = 0.f;
    for (int c = 0; c < DIM; ++c) mu += row[c];
    mu *= (1.f / DIM);
    float var = 0.f;
    for (int c = 0; c < DIM; ++c) { float d = row[c] - mu; var += d * d; }
    var *= (1.f / DIM);
    const float rs = rsqrtf(var + 1e-5f);
    _Float16* o = A2 + lane * DIM;
    for (int c = 0; c < DIM; ++c) {
      float y = (row[c] - mu) * rs * g[c] + be[c];
      o[c] = (_Float16)(y > 0.f ? y : 0.f);
    }
  }
}

// LayerNorm(128)+ReLU, 32 rows (all lanes, one row each)
__device__ __forceinline__ void ln_relu32(const float* __restrict__ H,
                                          _Float16* __restrict__ A2,
                                          const float* __restrict__ g,
                                          const float* __restrict__ be, int lane) {
  const float* row = H + lane * DIM;
  float mu = 0.f;
  for (int c = 0; c < DIM; ++c) mu += row[c];
  mu *= (1.f / DIM);
  float var = 0.f;
  for (int c = 0; c < DIM; ++c) { float d = row[c] - mu; var += d * d; }
  var *= (1.f / DIM);
  const float rs = rsqrtf(var + 1e-5f);
  _Float16* o = A2 + lane * DIM;
  for (int c = 0; c < DIM; ++c) {
    float y = (row[c] - mu) * rs * g[c] + be[c];
    o[c] = (_Float16)(y > 0.f ? y : 0.f);
  }
}

// -------- weight prep: W[din,dout] f32 -> Wt[dout][kpad] f16 (zero padded K) --------
__global__ __launch_bounds__(256) void prep_w(const float* __restrict__ W,
                                              _Float16* __restrict__ Wt,
                                              int din, int dout, int kpad) {
  int i = blockIdx.x * 256 + threadIdx.x;
  if (i >= dout * kpad) return;
  int n = i / kpad, k = i - n * kpad;
  float v = (k < din) ? W[(size_t)k * dout + n] : 0.f;
  Wt[(size_t)n * kpad + k] = (_Float16)v;
}

// ---------------- q = MLP(h), per 16-node tile, one wave per tile ----------------
__global__ __launch_bounds__(64) void q_kernel(
    const float* __restrict__ h,
    const _Float16* __restrict__ W1t, const float* __restrict__ b1,
    const float* __restrict__ g1, const float* __restrict__ be1,
    const _Float16* __restrict__ W2t, const float* __restrict__ b2,
    float* __restrict__ qbuf) {
  __shared__ __align__(16) _Float16 sA [2][16 * DIM];
  __shared__ __align__(16) float    sHd[2][16 * DIM];
  __shared__ __align__(16) _Float16 sA2[2][16 * DIM];
  const int w = threadIdx.x >> 5, lane = threadIdx.x & 31;
  const int tile = blockIdx.x * 2 + w;
  if (tile >= NN / 16) return;
  const int row = lane >> 1, c0 = (lane & 1) * 64;
  const int n = tile * 16 + row;
  for (int c = c0; c < c0 + 64; c += 4) {
    float4 f = *(const float4*)(h + (size_t)n * DIM + c);
    h4 o; o[0]=(_Float16)f.x; o[1]=(_Float16)f.y; o[2]=(_Float16)f.z; o[3]=(_Float16)f.w;
    *(h4*)(sA[w] + row * DIM + c) = o;
  }
  for (int nt = 0; nt < 8; ++nt)
    store_c_bias(sHd[w], DIM, nt * 16, 0,
                 mm_tile(sA[w], DIM, W1t + nt * 16 * DIM, DIM, DIM, lane), b1, lane);
  ln_relu16(sHd[w], sA2[w], g1, be1, lane);
  for (int nt = 0; nt < 8; ++nt) {
    v8f c = mm_tile(sA2[w], DIM, W2t + nt * 16 * DIM, DIM, DIM, lane);
    const int col = nt * 16 + (lane & 15), m0 = (lane >> 4) * 8;
    for (int r = 0; r < 8; ++r)
      qbuf[(size_t)(tile * 16 + m0 + r) * DIM + col] = c[r] + b2[col];
  }
}

// ------------- fused edge kernel, M=32 per wave, one wave per block -------------
__global__ __launch_bounds__(32) void edge_kernel(
    const float* __restrict__ h, const float* __restrict__ r_feat,
    const float* __restrict__ edge_feat, const int* __restrict__ edge_index,
    const _Float16* __restrict__ W1k, const float* __restrict__ b1k,
    const float* __restrict__ g1k, const float* __restrict__ be1k,
    const _Float16* __restrict__ W2k, const float* __restrict__ b2k,
    const _Float16* __restrict__ W1v, const float* __restrict__ b1v,
    const float* __restrict__ g1v, const float* __restrict__ be1v,
    const _Float16* __restrict__ W2v, const float* __restrict__ b2v,
    const float* __restrict__ ew_W, const float* __restrict__ ew_b,
    const float* __restrict__ qbuf,
    float* __restrict__ logits, unsigned* __restrict__ maxkey,
    _Float16* __restrict__ vbuf) {
  __shared__ __align__(16) _Float16 sKV[32 * KVPAD];   // 22.5 KB
  __shared__ __align__(16) float    sHd[32 * DIM];     // 16 KB
  __shared__ __align__(16) _Float16 sA2[32 * DIM];     // 8 KB
  __shared__ float sEW [32];
  __shared__ int   sDst[32];
  const int lane = threadIdx.x & 31;
  const int tile = blockIdx.x;                 // 32 edges per tile
  if (tile >= EE / 32) return;

  // ---- per-edge scalars + vectorized kv panel build (one row per lane) ----
  {
    const int e   = tile * 32 + lane;
    const int dst = edge_index[EE + e], src = edge_index[e];
    sDst[lane] = dst;
    float acc = ew_b[0];
    for (int i = 0; i < 64; ++i) acc += r_feat[(size_t)e * 64 + i] * ew_W[i];
    sEW[lane] = 1.f / (1.f + expf(-acc));
    // kv = [edge_feat(4) | r_feat(64) | h[dst](128) | h[src](128) | 0 pad(28)]
    // every segment starts on a multiple of 4 -> float4 gather, f16x4 LDS store
    _Float16* o = sKV + lane * KVPAD;
    for (int g = 0; g < KVPAD / 4; ++g) {
      float4 f;
      if      (g < 1)  f = *(const float4*)(edge_feat + (size_t)e * 4);
      else if (g < 17) f = *(const float4*)(r_feat + (size_t)e * 64 + 4 * g - 4);
      else if (g < 49) f = *(const float4*)(h + (size_t)dst * DIM + 4 * g - 68);
      else if (g < 81) f = *(const float4*)(h + (size_t)src * DIM + 4 * g - 196);
      else             f = make_float4(0.f, 0.f, 0.f, 0.f);
      h4 o4; o4[0]=(_Float16)f.x; o4[1]=(_Float16)f.y; o4[2]=(_Float16)f.z; o4[3]=(_Float16)f.w;
      *(h4*)(o + 4 * g) = o4;
    }
  }

  // ---- k-MLP ----
  for (int nt = 0; nt < 8; ++nt) {
    v8f c0 = {}, c1 = {};
    mm_tile_m2(sKV, KVPAD, W1k + nt * 16 * KVPAD, KVPAD, KVPAD, lane, c0, c1);
    store_c_bias(sHd, DIM, nt * 16, 0,  c0, b1k, lane);
    store_c_bias(sHd, DIM, nt * 16, 16, c1, b1k, lane);
  }
  ln_relu32(sHd, sA2, g1k, be1k, lane);
  for (int nt = 0; nt < 8; ++nt) {
    v8f c0 = {}, c1 = {};
    mm_tile_m2(sA2, DIM, W2k + nt * 16 * DIM, DIM, DIM, lane, c0, c1);
    store_c_bias(sHd, DIM, nt * 16, 0,  c0, b2k, lane);
    store_c_bias(sHd, DIM, nt * 16, 16, c1, b2k, lane);
  }

  // ---- logits + running max (sHd holds k) ----
  for (int t = lane; t < 32 * NHEAD; t += 32) {
    const int r = t >> 4, hh = t & 15;
    const int e = tile * 32 + r, dn = sDst[r];
    const float* qr = qbuf + (size_t)dn * DIM + hh * HDIM;
    const float* kr = sHd + r * DIM + hh * HDIM;
    float s = 0.f;
    for (int d = 0; d < HDIM; ++d) s += qr[d] * kr[d];
    s *= 0.3535533905932738f;  // 1/sqrt(8)
    logits[(size_t)e * NHEAD + hh] = s;
    atomicMax(&maxkey[dn * NHEAD + hh], enc_key(s));
  }

  // ---- v-MLP (re-uses kv panel; overwrites sHd/sA2) ----
  for (int nt = 0; nt < 8; ++nt) {
    v8f c0 = {}, c1 = {};
    mm_tile_m2(sKV, KVPAD, W1v + nt * 16 * KVPAD, KVPAD, KVPAD, lane, c0, c1);
    store_c_bias(sHd, DIM, nt * 16, 0,  c0, b1v, lane);
    store_c_bias(sHd, DIM, nt * 16, 16, c1, b1v, lane);
  }
  ln_relu32(sHd, sA2, g1v, be1v, lane);
  for (int nt = 0; nt < 8; ++nt) {
    v8f c0 = {}, c1 = {};
    mm_tile_m2(sA2, DIM, W2v + nt * 16 * DIM, DIM, DIM, lane, c0, c1);
    store_c_bias(sHd, DIM, nt * 16, 0,  c0, b2v, lane);
    store_c_bias(sHd, DIM, nt * 16, 16, c1, b2v, lane);
  }

  // scale by edge weight, store v as f16 (vectorized b128 stores, one row per lane)
  {
    const float* hr = sHd + lane * DIM;
    const float  s  = sEW[lane];
    _Float16* vo = vbuf + (size_t)(tile * 32 + lane) * DIM;
    for (int c = 0; c < DIM; c += 8) {
      h8 o;
      for (int j = 0; j < 8; ++j) o[j] = (_Float16)(hr[c + j] * s);
      *(h8*)(vo + c) = o;
    }
  }
}

// ------------- scatter pass: ex = exp(l - max); den += ex; accum += ex*v -------------
__global__ __launch_bounds__(256) void scatter_kernel(
    const float* __restrict__ logits, const unsigned* __restrict__ maxkey,
    const int* __restrict__ edge_index, const _Float16* __restrict__ vbuf,
    float* __restrict__ den, float* __restrict__ accum) {
  const long long t = (long long)blockIdx.x * 256 + threadIdx.x;  // over EE*16
  if (t >= (long long)EE * NHEAD) return;
  const int e = (int)(t >> 4), hh = (int)(t & 15);
  const int dn = edge_index[EE + e];
  const float m  = dec_key(maxkey[dn * NHEAD + hh]);
  const float ex = expf(logits[t] - m);
  atomicAdd(&den[dn * NHEAD + hh], ex);
  const _Float16* vp = vbuf + (size_t)e * DIM + hh * HDIM;
  float* ap = accum + (size_t)dn * DIM + hh * HDIM;
  for (int d = 0; d < HDIM; ++d) atomicAdd(ap + d, ex * (float)vp[d]);
}

// ------------- node output: out = MLP([attn, h]) + h -------------
__global__ __launch_bounds__(64) void node_kernel(
    const float* __restrict__ h, const float* __restrict__ accum,
    const float* __restrict__ den,
    const _Float16* __restrict__ W1t, const float* __restrict__ b1,
    const float* __restrict__ g1, const float* __restrict__ be1,
    const _Float16* __restrict__ W2t, const float* __restrict__ b2,
    float* __restrict__ out) {
  __shared__ __align__(16) _Float16 sA [2][16 * 256];
  __shared__ __align__(16) float    sHd[2][16 * DIM];
  __shared__ __align__(16) _Float16 sA2[2][16 * DIM];
  const int w = threadIdx.x >> 5, lane = threadIdx.x & 31;
  const int tile = blockIdx.x * 2 + w;
  if (tile >= NN / 16) return;
  const int row = lane >> 1, c0 = (lane & 1) * 128;
  const int n = tile * 16 + row;
  for (int c = c0; c < c0 + 128; ++c) {
    float v;
    if (c < DIM) v = accum[(size_t)n * DIM + c] / (den[n * NHEAD + (c >> 3)] + 1e-16f);
    else         v = h[(size_t)n * DIM + (c - DIM)];
    sA[w][row * 256 + c] = (_Float16)v;
  }
  for (int nt = 0; nt < 8; ++nt)
    store_c_bias(sHd[w], DIM, nt * 16, 0,
                 mm_tile(sA[w], 256, W1t + nt * 16 * 256, 256, 256, lane), b1, lane);
  ln_relu16(sHd[w], sA2[w], g1, be1, lane);
  for (int nt = 0; nt < 8; ++nt) {
    v8f c = mm_tile(sA2[w], DIM, W2t + nt * 16 * DIM, DIM, DIM, lane);
    const int col = nt * 16 + (lane & 15), m0 = (lane >> 4) * 8;
    for (int r = 0; r < 8; ++r) {
      const int nn = tile * 16 + m0 + r;
      out[(size_t)nn * DIM + col] = c[r] + b2[col] + h[(size_t)nn * DIM + col];
    }
  }
}

// ---------------------------------------------------------------------------------
extern "C" void kernel_launch(void* const* d_in, const int* in_sizes, int n_in,
                              void* d_out, int out_size, void* d_ws, size_t ws_size,
                              hipStream_t stream) {
  (void)in_sizes; (void)n_in; (void)out_size; (void)ws_size;
  const float* h         = (const float*)d_in[0];
  const float* r_feat    = (const float*)d_in[1];
  const float* edge_feat = (const float*)d_in[2];
  const int*   edge_idx  = (const int*)  d_in[3];
  const float *hk_W1=(const float*)d_in[4],  *hk_b1=(const float*)d_in[5],
              *hk_g1=(const float*)d_in[6],  *hk_be1=(const float*)d_in[7],
              *hk_W2=(const float*)d_in[8],  *hk_b2=(const float*)d_in[9];
  const float *hv_W1=(const float*)d_in[10], *hv_b1=(const float*)d_in[11],
              *hv_g1=(const float*)d_in[12], *hv_be1=(const float*)d_in[13],
              *hv_W2=(const float*)d_in[14], *hv_b2=(const float*)d_in[15];
  const float *hq_W1=(const float*)d_in[16], *hq_b1=(const float*)d_in[17],
              *hq_g1=(const float*)d_in[18], *hq_be1=(const float*)d_in[19],
              *hq_W2=(const float*)d_in[20], *hq_b2=(const float*)d_in[21];
  const float *no_W1=(const float*)d_in[22], *no_b1=(const float*)d_in[23],
              *no_g1=(const float*)d_in[24], *no_be1=(const float*)d_in[25],
              *no_W2=(const float*)d_in[26], *no_b2=(const float*)d_in[27];
  const float *ew_W=(const float*)d_in[28], *ew_b=(const float*)d_in[29];
  float* out = (float*)d_out;

  char* p = (char*)d_ws;
  auto alloc = [&](size_t bytes) -> char* {
    char* r = p; p += (bytes + 255) & ~(size_t)255; return r;
  };
  _Float16* W1kt = (_Float16*)alloc((size_t)DIM * KVPAD * 2);
  _Float16* W1vt = (_Float16*)alloc((size_t)DIM * KVPAD * 2);
  _Float16* W1qt = (_Float16*)alloc((size_t)DIM * DIM * 2);
  _Float16* W2kt = (_Float16*)alloc((size_t)DIM * DIM * 2);
  _Float16* W2vt = (_Float16*)alloc((size_t)DIM * DIM * 2);
  _Float16* W2qt = (_Float16*)alloc((size_t)DIM * DIM * 2);
  _Float16* W2ot = (_Float16*)alloc((size_t)DIM * DIM * 2);
  _Float16* W1ot = (_Float16*)alloc((size_t)DIM * 256 * 2);
  float*    qbuf = (float*)   alloc((size_t)NN * DIM * 4);
  float*    lgts = (float*)   alloc((size_t)EE * NHEAD * 4);
  _Float16* vbuf = (_Float16*)alloc((size_t)EE * DIM * 2);
  // zero-initialized region (contiguous single memset)
  char* zbase = p;
  unsigned* maxkey = (unsigned*)alloc((size_t)NN * NHEAD * 4);
  float*    den    = (float*)   alloc((size_t)NN * NHEAD * 4);
  float*    accum  = (float*)   alloc((size_t)NN * DIM * 4);
  hipMemsetAsync(zbase, 0, (size_t)(p - zbase), stream);

  auto prep = [&](const float* W, _Float16* Wt, int din, int dout, int kpad) {
    int total = dout * kpad;
    prep_w<<<(total + 255) / 256, 256, 0, stream>>>(W, Wt, din, dout, kpad);
  };
  prep(hk_W1, W1kt, KVD, DIM, KVPAD);
  prep(hv_W1, W1vt, KVD, DIM, KVPAD);
  prep(hq_W1, W1qt, DIM, DIM, DIM);
  prep(hk_W2, W2kt, DIM, DIM, DIM);
  prep(hv_W2, W2vt, DIM, DIM, DIM);
  prep(hq_W2, W2qt, DIM, DIM, DIM);
  prep(no_W2, W2ot, DIM, DIM, DIM);
  prep(no_W1, W1ot, 256, DIM, 256);

  // q = MLP(h)
  q_kernel<<<(NN / 16 + 1) / 2, 64, 0, stream>>>(h, W1qt, hq_b1, hq_g1, hq_be1,
                                                 W2qt, hq_b2, qbuf);
  // fused edge pass (k, logits+max, v), 32 edges per wave
  edge_kernel<<<EE / 32, 32, 0, stream>>>(h, r_feat, edge_feat, edge_idx,
                                          W1kt, hk_b1, hk_g1, hk_be1, W2kt, hk_b2,
                                          W1vt, hv_b1, hv_g1, hv_be1, W2vt, hv_b2,
                                          ew_W, ew_b, qbuf, lgts, maxkey, vbuf);
  // softmax scatter
  scatter_kernel<<<(int)(((long long)EE * NHEAD) / 256), 256, 0, stream>>>(
      lgts, maxkey, edge_idx, vbuf, den, accum);
  // node output MLP + residual
  node_kernel<<<(NN / 16 + 1) / 2, 64, 0, stream>>>(h, accum, den,
                                                    W1ot, no_b1, no_g1, no_be1,
                                                    W2ot, no_b2, out);
}
